// LogSpacePolynomialCell_27066883900118
// MI455X (gfx1250) — matrix operand order repaired and checked by reading, not compiled, not run on hardware
//
#include <hip/hip_runtime.h>
#include <hip/hip_bf16.h>

// ---------------------------------------------------------------------------
// LogSpacePolynomialCell for MI455X (gfx1250, wave32, WMMA)
//
// Phase 1: three GEMMs  out[m,e] = sum_d X[m,d] * W[e,d]
//   M = T*B = 8192, K = N = D = 1024.  bf16 hi/lo split (x = hi + lo) with
//   v_wmma_f32_16x16x32_bf16, 3 products (hh + hl + lh) ~= fp32 accuracy.
// Phase 2: elementwise transforms + T=512 sequential scan, parallel over
//   B*D = 16384 independent recurrences.
// ---------------------------------------------------------------------------

typedef __attribute__((ext_vector_type(16))) __bf16 v16bf;
typedef __attribute__((ext_vector_type(8)))  float  v8f;

#define TT 512
#define BB 16
#define DD 1024
#define MM (TT * BB)          // 8192 rows of flattened x
#define KK DD
#define NN DD
#define TBDELEM ((size_t)TT * BB * DD)   // 8,388,608 elements per tensor

#define LOG_ZERO (-40.0f)

// LDS row stride in bf16 elements: 32 data + 8 pad = 80 bytes.
// 16B aligned rows; 20*r mod 64 is distinct for r=0..15 -> conflict-free
// 16-lane row-parallel b128 fragment loads.
#define LROW 40

// ---------------------------------------------------------------------------
// GEMM: grid = (N/64, M/128, 3); z selects which W / output buffer.
// Block 256 threads = 8 waves. Wave w covers rows [ (w>>2)*64, +64 ) as four
// 16x16 subtiles, cols [ (w&3)*16, +16 ).
// ---------------------------------------------------------------------------
__global__ __launch_bounds__(256)
void lspc_gemm3_bf16x3(const float* __restrict__ X,
                       const float* __restrict__ W0,
                       const float* __restrict__ W1,
                       const float* __restrict__ W2,
                       float* __restrict__ O0,
                       float* __restrict__ O1,
                       float* __restrict__ O2)
{
    __shared__ __bf16 sXhi[128 * LROW];
    __shared__ __bf16 sXlo[128 * LROW];
    __shared__ __bf16 sWhi[64 * LROW];
    __shared__ __bf16 sWlo[64 * LROW];

    const float* __restrict__ W = (blockIdx.z == 0) ? W0 : ((blockIdx.z == 1) ? W1 : W2);
    float* __restrict__       O = (blockIdx.z == 0) ? O0 : ((blockIdx.z == 1) ? O1 : O2);

    const int tid    = threadIdx.x;
    const int lane   = tid & 31;
    const int wave   = tid >> 5;
    const int khalf  = lane >> 4;     // 0: lanes 0-15, 1: lanes 16-31
    const int lanelo = lane & 15;
    const int n0     = (wave & 3) * 16;
    const int m0     = (wave >> 2) * 64;
    const int mBase  = blockIdx.y * 128;
    const int nBase  = blockIdx.x * 64;

    v8f acc[4];
    #pragma unroll
    for (int s = 0; s < 4; ++s) acc[s] = (v8f){0.f, 0.f, 0.f, 0.f, 0.f, 0.f, 0.f, 0.f};

    union Frag { v16bf v; uint4 q[2]; };

    for (int kb = 0; kb < KK; kb += 32) {
        // ---- stage X tile: 128 rows x 32 k (fp32 -> bf16 hi/lo) ----
        #pragma unroll
        for (int it = 0; it < 4; ++it) {
            int idx = tid + it * 256;              // 1024 float4s
            int r   = idx >> 3;
            int c4  = (idx & 7) << 2;
            const float4 v = *(const float4*)&X[(size_t)(mBase + r) * KK + kb + c4];
            float f[4] = {v.x, v.y, v.z, v.w};
            #pragma unroll
            for (int j = 0; j < 4; ++j) {
                __bf16 h = (__bf16)f[j];
                sXhi[r * LROW + c4 + j] = h;
                sXlo[r * LROW + c4 + j] = (__bf16)(f[j] - (float)h);
            }
        }
        // ---- stage W tile: 64 rows(n) x 32 k ----
        #pragma unroll
        for (int it = 0; it < 2; ++it) {
            int idx = tid + it * 256;              // 512 float4s
            int r   = idx >> 3;
            int c4  = (idx & 7) << 2;
            const float4 v = *(const float4*)&W[(size_t)(nBase + r) * KK + kb + c4];
            float f[4] = {v.x, v.y, v.z, v.w};
            #pragma unroll
            for (int j = 0; j < 4; ++j) {
                __bf16 h = (__bf16)f[j];
                sWhi[r * LROW + c4 + j] = h;
                sWlo[r * LROW + c4 + j] = (__bf16)(f[j] - (float)h);
            }
        }
        __syncthreads();

        // ---- B fragments (32x16 bf16): lane holds N = n0+lanelo,
        //      K chunks [8*khalf,+8) in v0..3 and [16+8*khalf,+8) in v4..7 ----
        Frag bhi, blo;
        {
            const char* rowH = (const char*)&sWhi[(n0 + lanelo) * LROW];
            const char* rowL = (const char*)&sWlo[(n0 + lanelo) * LROW];
            const int off = 16 * khalf;            // bytes
            bhi.q[0] = *(const uint4*)(rowH + off);
            bhi.q[1] = *(const uint4*)(rowH + off + 32);
            blo.q[0] = *(const uint4*)(rowL + off);
            blo.q[1] = *(const uint4*)(rowL + off + 32);
        }

        // ---- per M-subtile: A fragments + 3 WMMAs (hh, hl, lh) ----
        #pragma unroll
        for (int s = 0; s < 4; ++s) {
            // A (16x32 bf16): lane holds M = row, K chunks [8*khalf,+8) in
            // v0..3 and [16+8*khalf,+8) in v4..7.
            Frag ahi, alo;
            const char* rowH = (const char*)&sXhi[(m0 + 16 * s + lanelo) * LROW];
            const char* rowL = (const char*)&sXlo[(m0 + 16 * s + lanelo) * LROW];
            const int off0 = 16 * khalf;           // bytes
            ahi.q[0] = *(const uint4*)(rowH + off0);
            ahi.q[1] = *(const uint4*)(rowH + 32 + off0);
            alo.q[0] = *(const uint4*)(rowL + off0);
            alo.q[1] = *(const uint4*)(rowL + 32 + off0);

            acc[s] = __builtin_amdgcn_wmma_f32_16x16x32_bf16(
                         false, ahi.v, false, bhi.v, (short)0, acc[s], false, false);
            acc[s] = __builtin_amdgcn_wmma_f32_16x16x32_bf16(
                         false, ahi.v, false, blo.v, (short)0, acc[s], false, false);
            acc[s] = __builtin_amdgcn_wmma_f32_16x16x32_bf16(
                         false, alo.v, false, bhi.v, (short)0, acc[s], false, false);
        }
        __syncthreads();
    }

    // ---- epilogue: C/D layout VGPR j -> M = j + 8*khalf, N = lanelo ----
    #pragma unroll
    for (int s = 0; s < 4; ++s) {
        const int mg = mBase + m0 + 16 * s + 8 * khalf;
        const int ng = nBase + n0 + lanelo;
        #pragma unroll
        for (int j = 0; j < 8; ++j)
            O[(size_t)(mg + j) * NN + ng] = acc[s][j];
    }
}

// ---------------------------------------------------------------------------
// Scan: one thread per (b,d) recurrence.
// ---------------------------------------------------------------------------
__device__ __forceinline__ float softplusf(float x) {
    return fmaxf(x, 0.f) + log1pf(__expf(-fabsf(x)));
}

__device__ __forceinline__ void signed_log_add(float la, float sa,
                                               float lb, float sb,
                                               float& lr, float& sr)
{
    const bool az = la <= LOG_ZERO + 1.0f;
    const bool bz = lb <= LOG_ZERO + 1.0f;
    const float mx = fmaxf(la, lb);
    const float mn = fminf(la, lb);
    const bool amax = la >= lb;
    const float smax = amax ? sa : sb;
    const float smin = amax ? sb : sa;
    const bool same = (smax * smin) > 0.f;
    const float ed = __expf(mn - mx);
    const float lsame = mx + log1pf(ed);
    const float lopp  = mx + log1pf(-fminf(ed, 0.9999f));
    float l = same ? lsame : lopp;
    float s = smax;
    if (az)      { l = lb; s = sb; }
    else if (bz) { l = la; s = sa; }
    lr = l; sr = s;
}

__global__ __launch_bounds__(256)
void lspc_scan(const float* __restrict__ lin_raw,
               const float* __restrict__ alpha_raw,
               const float* __restrict__ delta_raw,
               const float* __restrict__ b,
               const float* __restrict__ b_alpha,
               const float* __restrict__ b_delta,
               const float* __restrict__ log_r_h,
               const float* __restrict__ sign_r_h,
               float* __restrict__ out)
{
    const int bd = blockIdx.x * 256 + threadIdx.x;   // [0, B*D)
    const int d  = bd & (DD - 1);

    const float bx  = b[d];
    const float ba  = b_alpha[d];
    const float bdl = b_delta[d];
    const float lrh = log_r_h[d];
    const float srh = sign_r_h[d];

    float* __restrict__ log_h  = out;                                  // [T+1,B,D]
    float* __restrict__ sign_h = out + (size_t)(TT + 1) * BB * DD;     // [T+1,B,D]
    float* __restrict__ h_lin  = sign_h + (size_t)(TT + 1) * BB * DD;  // [T,B,D]

    float lhp = LOG_ZERO, shp = 1.0f;
    log_h[bd]  = lhp;     // t = 0 slice
    sign_h[bd] = shp;

    const int stride = BB * DD;
    for (int t = 0; t < TT; ++t) {
        const size_t off = (size_t)t * stride + bd;

        // alpha = 1 + softplus(x@W_alpha + b_alpha)
        const float alpha = 1.0f + softplusf(alpha_raw[off] + ba);

        // to_log_space(x@W_x + b)
        const float lin = lin_raw[off] + bx;
        const float sin_ = (lin < 0.f) ? -1.f : 1.f;
        const float axl  = fabsf(lin);
        const float llin = (axl > 1e-10f) ? __logf(axl) : LOG_ZERO;

        // log_sigmoid(+-(x@W_delta + b_delta))
        const float dr = delta_raw[off] + bdl;
        const float lcm = log1pf(__expf(-fabsf(dr)));
        const float log_d   = fminf(dr, 0.f)  - lcm;   // log sigmoid(dr)
        const float log_1md = fminf(-dr, 0.f) - lcm;   // log sigmoid(-dr)

        // v = r_h * h_prev + input   (signed log space)
        float lv, sv;
        signed_log_add(lrh + lhp, srh * shp, llin, sin_, lv, sv);

        // candidate = soft_bound_log(alpha * log_v)
        const float log_cand = -softplusf(-(alpha * lv));

        // h = (1-delta)*h_prev + delta*cand   (signed log space)
        float lh, sh;
        signed_log_add(log_1md + lhp, shp, log_d + log_cand, sv, lh, sh);

        // from_log_space (forward: grad_scale is identity)
        const float res = sh * __expf(fminf(lh, 20.0f));
        const float hl  = (lh > LOG_ZERO + 1.0f) ? res : 0.0f;

        const size_t off1 = (size_t)(t + 1) * stride + bd;
        log_h[off1]  = lh;
        sign_h[off1] = sh;
        h_lin[off]   = hl;

        lhp = lh; shp = sh;
    }
}

// ---------------------------------------------------------------------------
extern "C" void kernel_launch(void* const* d_in, const int* in_sizes, int n_in,
                              void* d_out, int out_size, void* d_ws, size_t ws_size,
                              hipStream_t stream)
{
    const float* x        = (const float*)d_in[0];
    const float* W_x      = (const float*)d_in[1];
    const float* W_alpha  = (const float*)d_in[2];
    const float* W_delta  = (const float*)d_in[3];
    const float* b        = (const float*)d_in[4];
    const float* b_alpha  = (const float*)d_in[5];
    const float* b_delta  = (const float*)d_in[6];
    const float* log_r_h  = (const float*)d_in[7];
    const float* sign_r_h = (const float*)d_in[8];

    float* out = (float*)d_out;
    float* ws  = (float*)d_ws;
    float* lin_raw   = ws;
    float* alpha_raw = ws + TBDELEM;
    float* delta_raw = ws + 2 * TBDELEM;

    dim3 gGrid(NN / 64, MM / 128, 3);   // (16, 64, 3)
    lspc_gemm3_bf16x3<<<gGrid, 256, 0, stream>>>(
        x, W_x, W_alpha, W_delta, lin_raw, alpha_raw, delta_raw);

    lspc_scan<<<dim3((BB * DD) / 256), 256, 0, stream>>>(
        lin_raw, alpha_raw, delta_raw, b, b_alpha, b_delta,
        log_r_h, sign_r_h, out);
}
